// NNPolicy_88021059764292
// MI455X (gfx1250) — matrix operand order repaired and checked by reading, not compiled, not run on hardware
//
#include <hip/hip_runtime.h>
#include <hip/hip_bf16.h>

typedef __attribute__((ext_vector_type(16))) __bf16 bf16x16;
typedef __attribute__((ext_vector_type(4)))  __bf16 bf16x4;
typedef __attribute__((ext_vector_type(8)))  float  f32x8;
typedef __attribute__((ext_vector_type(4)))  int    i32x4;

#define M_TOTAL 512
#define K_DIM   256
#define N_TOTAL 262144
#define TOPK    16
#define LAB_DIM 112

#define MTILE   128                   // obs rows per block
#define NSLAB   2048                  // db rows per block
#define NTILE   64                    // inner db tile
#define NTILES  (NSLAB / NTILE)       // 32
#define NSLABS  (N_TOTAL / NSLAB)     // 128
#define MBLOCKS (M_TOTAL / MTILE)     // 4

#define DB_STRIDE  272                // padded bf16 row stride (544B: conflict-free)
#define D2_STRIDE  65                 // padded f32 row stride
#define STG_STRIDE 260                // padded fp32 staging row stride

#define BIGF 3.402823466e38f

// CDNA5 async global->LDS path (ASYNCcnt-tracked), with compile-safe fallback.
// Probe round 2 revealed the builtin's signature: (int4 as1*, int4 as3*, Ii, Ii).
#if __has_builtin(__builtin_amdgcn_global_load_async_to_lds_b128) && \
    __has_builtin(__builtin_amdgcn_s_wait_asynccnt)
#define USE_ASYNC_LDS 1
typedef __attribute__((address_space(1))) i32x4* gas1_ptr;
typedef __attribute__((address_space(3))) i32x4* las3_ptr;
#else
#define USE_ASYNC_LDS 0
#endif

__device__ __forceinline__ void topk_insert(float (&bd)[TOPK], int (&bi)[TOPK],
                                            float d, int g) {
    if (d < bd[TOPK - 1]) {
        bd[TOPK - 1] = d; bi[TOPK - 1] = g;
        #pragma unroll
        for (int s = TOPK - 1; s > 0; --s) {
            if (bd[s] < bd[s - 1]) {
                float td = bd[s]; bd[s] = bd[s - 1]; bd[s - 1] = td;
                int   ti = bi[s]; bi[s] = bi[s - 1]; bi[s - 1] = ti;
            }
        }
    }
}

// ---------------------------------------------------------------------------
// Phase 1: bf16x3 WMMA distance tiles + streaming per-row top-16 per N-slab.
// grid = (MBLOCKS=4 fastest, NSLABS=128): the 4 M-blocks of a slab run
// together and share the db slab through L2 (db streams HBM once overall).
// ---------------------------------------------------------------------------
__global__ __launch_bounds__(256)
void nn_gemm_topk(const float* __restrict__ obs,
                  const float* __restrict__ db,
                  float* __restrict__ part_d2,
                  int*   __restrict__ part_idx)
{
    extern __shared__ char smem_raw[];
    __bf16* db_hi = (__bf16*)smem_raw;                  // 64 x 272 bf16
    __bf16* db_lo = db_hi + NTILE * DB_STRIDE;          // 64 x 272 bf16
    float*  d2s   = (float*)(db_lo + NTILE * DB_STRIDE);// 128 x 65 f32
    float*  ynorm = d2s + MTILE * D2_STRIDE;            // 64 f32
    float*  xnorm = ynorm + NTILE;                      // 128 f32
    float*  stage = xnorm + MTILE;                      // 2 x 64 x 260 f32 (async)

    const int tid  = threadIdx.x;
    const int lane = tid & 31;
    const int wave = tid >> 5;       // 0..7, each wave owns 16 obs rows
    const int mb   = blockIdx.x;     // 0..3
    const int slab = blockIdx.y;     // 0..127

    const int srow = tid >> 2;       // staging: 4 threads per db row
    const int scol = (tid & 3) * 64;

#if USE_ASYNC_LDS
    // Kick off tile 0: per-thread 16x B128 async copies (own region only).
    {
        const float* g0 = db + (size_t)(slab * NSLAB + srow) * K_DIM + scol;
        float* l0 = stage + srow * STG_STRIDE + scol;
        #pragma unroll
        for (int v = 0; v < 16; ++v)
            __builtin_amdgcn_global_load_async_to_lds_b128(
                (gas1_ptr)(g0 + v * 4), (las3_ptr)(l0 + v * 4), 0, 0);
    }
#endif

    // ---- A fragments: obs rows -> bf16 hi/lo, resident in VGPRs ----------
    // 16-bit A 16x32 layout: lane L holds row m=L%16; per k-step kb:
    // k = kb + (L/16)*16 + {0..15} (contiguous order chosen here; B matches)
    const int mrow = mb * MTILE + wave * 16 + (lane & 15);
    const int h8   = lane >> 4;
    bf16x16 a_hi[8], a_lo[8];
    float sq = 0.0f;
    #pragma unroll
    for (int t = 0; t < 8; ++t) {
        const float* p0 = obs + (size_t)mrow * K_DIM + t * 32 + h8 * 16;
        #pragma unroll
        for (int j = 0; j < 8; ++j) {
            float x0 = p0[j];
            float x1 = p0[8 + j];
            __bf16 h0 = (__bf16)x0, h1 = (__bf16)x1;
            a_hi[t][j]     = h0;  a_lo[t][j]     = (__bf16)(x0 - (float)h0);
            a_hi[t][8 + j] = h1;  a_lo[t][8 + j] = (__bf16)(x1 - (float)h1);
            sq += x0 * x0 + x1 * x1;
        }
    }
    sq += __shfl_xor(sq, 16, 32);          // combine the two k-half lanes
    if (lane < 16) xnorm[wave * 16 + lane] = sq;

    // ---- running top-16 (lanes 0..127 each own one obs row) --------------
    float bd[TOPK]; int bi[TOPK];
    #pragma unroll
    for (int k = 0; k < TOPK; ++k) { bd[k] = BIGF; bi[k] = 0; }

    for (int it = 0; it < NTILES; ++it) {
        const int n0 = slab * NSLAB + it * NTILE;
        __syncthreads();                   // staging vs previous selection

#if USE_ASYNC_LDS
        // Issue tile it+1 into the other buffer, then drain tile it.
        if (it + 1 < NTILES) {
            const float* g1 = db + (size_t)(n0 + NTILE + srow) * K_DIM + scol;
            float* l1 = stage + ((it + 1) & 1) * (NTILE * STG_STRIDE)
                              + srow * STG_STRIDE + scol;
            #pragma unroll
            for (int v = 0; v < 16; ++v)
                __builtin_amdgcn_global_load_async_to_lds_b128(
                    (gas1_ptr)(g1 + v * 4), (las3_ptr)(l1 + v * 4), 0, 0);
            __builtin_amdgcn_s_wait_asynccnt(16);  // tile it complete (in-order)
        } else {
            __builtin_amdgcn_s_wait_asynccnt(0);
        }
#endif

        // ---- convert db tile: fp32 -> bf16 hi/lo in LDS, fused |y|^2 -----
        {
#if USE_ASYNC_LDS
            const float* src = stage + (it & 1) * (NTILE * STG_STRIDE)
                                     + srow * STG_STRIDE + scol;
#else
            const float* src = db + (size_t)(n0 + srow) * K_DIM + scol;
            if (it + 1 < NTILES) {         // CDNA5 global_prefetch_b8
                __builtin_prefetch(src + NTILE * K_DIM, 0, 1);
                __builtin_prefetch(src + NTILE * K_DIM + 32, 0, 1);
            }
#endif
            float s = 0.0f;
            #pragma unroll
            for (int v = 0; v < 16; ++v) {
                float4 x = ((const float4*)src)[v];
                bf16x4 hv, lv;
                __bf16 h;
                h = (__bf16)x.x; hv[0] = h; lv[0] = (__bf16)(x.x - (float)h);
                h = (__bf16)x.y; hv[1] = h; lv[1] = (__bf16)(x.y - (float)h);
                h = (__bf16)x.z; hv[2] = h; lv[2] = (__bf16)(x.z - (float)h);
                h = (__bf16)x.w; hv[3] = h; lv[3] = (__bf16)(x.w - (float)h);
                const int c = scol + v * 4;
                *(bf16x4*)(db_hi + srow * DB_STRIDE + c) = hv;
                *(bf16x4*)(db_lo + srow * DB_STRIDE + c) = lv;
                s += x.x * x.x + x.y * x.y + x.z * x.z + x.w * x.w;
            }
            s += __shfl_xor(s, 1, 32);
            s += __shfl_xor(s, 2, 32);
            if ((tid & 3) == 0) ynorm[srow] = s;
        }
        __syncthreads();

        // ---- dot = obs . db via bf16x3; 4 independent accumulator chains -
        f32x8 acc[4];
        #pragma unroll
        for (int nf = 0; nf < 4; ++nf) {
            f32x8 z = {0.f, 0.f, 0.f, 0.f, 0.f, 0.f, 0.f, 0.f};
            acc[nf] = z;
        }
        #pragma unroll
        for (int t = 0; t < 8; ++t) {
            // B 32x16 operand, this kernel's k order: lane L holds col n=L%16,
            // k = kb + (L/16)*16 + {0..15} (contiguous in LDS)
            const int ko = t * 32 + h8 * 16;
            bf16x16 bh[4], bl[4];
            #pragma unroll
            for (int nf = 0; nf < 4; ++nf) {
                const int nl = nf * 16 + (lane & 15);
                bh[nf] = *(const bf16x16*)(db_hi + nl * DB_STRIDE + ko);
                bl[nf] = *(const bf16x16*)(db_lo + nl * DB_STRIDE + ko);
            }
            // Round-robin the 4 accumulators so same-acc WMMAs sit >=4 apart
            // (covers the bf16 WMMA->WMMA RAW hazard, ISA 7.12.1).
            #pragma unroll
            for (int nf = 0; nf < 4; ++nf)
                acc[nf] = __builtin_amdgcn_wmma_f32_16x16x32_bf16(
                    false, a_hi[t], false, bh[nf], (short)0, acc[nf], false, false);
            #pragma unroll
            for (int nf = 0; nf < 4; ++nf)
                acc[nf] = __builtin_amdgcn_wmma_f32_16x16x32_bf16(
                    false, a_hi[t], false, bl[nf], (short)0, acc[nf], false, false);
            #pragma unroll
            for (int nf = 0; nf < 4; ++nf)
                acc[nf] = __builtin_amdgcn_wmma_f32_16x16x32_bf16(
                    false, a_lo[t], false, bh[nf], (short)0, acc[nf], false, false);
        }
        // C/D layout: VGPR v: lanes 0-15 -> M=v, lanes 16-31 -> M=v+8
        #pragma unroll
        for (int nf = 0; nf < 4; ++nf) {
            const int nl    = nf * 16 + (lane & 15);
            const int mbase = wave * 16 + h8 * 8;
            #pragma unroll
            for (int v = 0; v < 8; ++v)
                d2s[(mbase + v) * D2_STRIDE + nl] = acc[nf][v];
        }
        __syncthreads();

        // ---- selection: d2 = |x|^2 + |y|^2 - 2*dot, keep 16 smallest -----
        if (tid < MTILE) {
            const float xn = xnorm[tid];
            for (int n = 0; n < NTILE; ++n) {
                float d2 = xn + ynorm[n] - 2.0f * d2s[tid * D2_STRIDE + n];
                topk_insert(bd, bi, d2, n0 + n);
            }
        }
    }

    if (tid < MTILE) {
        const int mg = mb * MTILE + tid;
        float* pd = part_d2  + ((size_t)mg * NSLABS + slab) * TOPK;
        int*   pi = part_idx + ((size_t)mg * NSLABS + slab) * TOPK;
        #pragma unroll
        for (int k = 0; k < TOPK; ++k) { pd[k] = bd[k]; pi[k] = bi[k]; }
    }
}

// ---------------------------------------------------------------------------
// Phase 2: merge 128 partial top-16 lists per row -> final top-16 + sqrt.
// ---------------------------------------------------------------------------
__global__ __launch_bounds__(32)
void nn_merge(const float* __restrict__ part_d2, const int* __restrict__ part_idx,
              float* __restrict__ fdist, int* __restrict__ fidx)
{
    __shared__ float sd[32 * TOPK];
    __shared__ int   si[32 * TOPK];
    const int b = blockIdx.x, lane = threadIdx.x;

    float bd[TOPK]; int bi[TOPK];
    #pragma unroll
    for (int k = 0; k < TOPK; ++k) { bd[k] = BIGF; bi[k] = 0; }

    const float* pd = part_d2  + (size_t)b * NSLABS * TOPK;
    const int*   pi = part_idx + (size_t)b * NSLABS * TOPK;
    for (int e = lane; e < NSLABS * TOPK; e += 32)
        topk_insert(bd, bi, pd[e], pi[e]);

    #pragma unroll
    for (int k = 0; k < TOPK; ++k) { sd[lane * TOPK + k] = bd[k]; si[lane * TOPK + k] = bi[k]; }
    __syncthreads();

    if (lane == 0) {
        #pragma unroll
        for (int k = 0; k < TOPK; ++k) { bd[k] = BIGF; bi[k] = 0; }
        for (int e = 0; e < 32 * TOPK; ++e)
            topk_insert(bd, bi, sd[e], si[e]);
        #pragma unroll
        for (int k = 0; k < TOPK; ++k) {
            fdist[b * TOPK + k] = sqrtf(fmaxf(bd[k], 0.0f));
            fidx [b * TOPK + k] = bi[k];
        }
    }
}

// ---------------------------------------------------------------------------
// Phase 3a: deterministic global softmax denominator (fixed-order reduction).
// ---------------------------------------------------------------------------
__global__ __launch_bounds__(256)
void nn_gsum(const float* __restrict__ fdist, float* __restrict__ gsum)
{
    __shared__ float red[256];
    const int t = threadIdx.x;
    float s = 0.0f;
    for (int i = t; i < M_TOTAL * TOPK; i += 256) s += expf(-fdist[i]);
    red[t] = s;
    __syncthreads();
    for (int off = 128; off; off >>= 1) {
        if (t < off) red[t] += red[t + off];
        __syncthreads();
    }
    if (t == 0) *gsum = red[0];
}

// ---------------------------------------------------------------------------
// Phase 3b: weighted label gather: out[b,:] = sum_k w_bk * labels[idx_bk,:]
// ---------------------------------------------------------------------------
__global__ __launch_bounds__(128)
void nn_out(const float* __restrict__ fdist, const int* __restrict__ fidx,
            const float* __restrict__ labels, const float* __restrict__ gsum,
            float* __restrict__ out)
{
    const int b = blockIdx.x, j = threadIdx.x;
    if (j >= LAB_DIM) return;
    const float inv = 1.0f / *gsum;
    float acc = 0.0f;
    #pragma unroll
    for (int k = 0; k < TOPK; ++k) {
        const float w = expf(-fdist[b * TOPK + k]) * inv;
        acc += w * labels[(size_t)fidx[b * TOPK + k] * LAB_DIM + j];
    }
    out[b * LAB_DIM + j] = acc;
}

extern "C" void kernel_launch(void* const* d_in, const int* in_sizes, int n_in,
                              void* d_out, int out_size, void* d_ws, size_t ws_size,
                              hipStream_t stream)
{
    const float* obs = (const float*)d_in[0];   // 512*4*64 f32
    const float* db  = (const float*)d_in[1];   // 262144*256 f32
    const float* lab = (const float*)d_in[2];   // 262144*112 f32
    // d_in[3] = topk (16): compile-time constant here.

    char* ws = (char*)d_ws;                     // needs ~8.1 MB
    float* part_d2  = (float*)ws; ws += (size_t)M_TOTAL * NSLABS * TOPK * sizeof(float);
    int*   part_idx = (int*)  ws; ws += (size_t)M_TOTAL * NSLABS * TOPK * sizeof(int);
    float* fdist    = (float*)ws; ws += (size_t)M_TOTAL * TOPK * sizeof(float);
    int*   fidx     = (int*)  ws; ws += (size_t)M_TOTAL * TOPK * sizeof(int);
    float* gsum     = (float*)ws;

    // Always reserve the async staging region so host/device layouts agree.
    const size_t smem = (size_t)2 * NTILE * DB_STRIDE * sizeof(__bf16)    // db hi+lo
                      + (size_t)MTILE * D2_STRIDE * sizeof(float)         // dot tile
                      + (NTILE + MTILE) * sizeof(float)                   // norms
                      + (size_t)2 * NTILE * STG_STRIDE * sizeof(float);   // async stage

    nn_gemm_topk<<<dim3(MBLOCKS, NSLABS), 256, smem, stream>>>(obs, db, part_d2, part_idx);
    nn_merge<<<dim3(M_TOTAL), 32, 0, stream>>>(part_d2, part_idx, fdist, fidx);
    nn_gsum<<<1, 256, 0, stream>>>(fdist, gsum);
    nn_out<<<dim3(M_TOTAL), 128, 0, stream>>>(fdist, fidx, lab, gsum, (float*)d_out);
}
